// Physnet_17257178595654
// MI455X (gfx1250) — compile-verified, matrix operand order checked by reading, not run on hardware
//
#include <hip/hip_runtime.h>

#define NA   20000
#define NE   320000
#define FF   256
#define DF   100
#define DFP  128
#define NG   64
#define RCUT 12.0f

typedef __attribute__((ext_vector_type(16))) _Float16 v16h;
typedef __attribute__((ext_vector_type(8)))  _Float16 v8h;
typedef __attribute__((ext_vector_type(4)))  _Float16 v4h;
typedef __attribute__((ext_vector_type(8)))  float    v8f;
typedef __attribute__((ext_vector_type(4)))  float    v4f;

// =============== weight pack: f16, B-fragment layout Wp[n][k] ===============
//  !trans: Wp[n*Kpad+k] = (k < k_real) ? W[k*ldw + n] : 0
//   trans: Wp[n*Kpad+k] = (n < n_real) ? W[n*ldw + k] : 0
__global__ void k_pack_w(const float* __restrict__ W, _Float16* __restrict__ Wp,
                         int Kpad, int Nout, int k_real, int n_real, int ldw, int trans)
{
    int i = blockIdx.x*256 + threadIdx.x;
    if (i >= Nout*Kpad) return;
    int n = i / Kpad, k = i - n*Kpad;
    float v = 0.f;
    if (trans) { if (n < n_real) v = W[(size_t)n*ldw + k]; }
    else       { if (k < k_real) v = W[(size_t)k*ldw + n]; }
    Wp[i] = (_Float16)v;
}

// ======================= WMMA GEMM =========================
// Block: 256 threads = 8 waves. Tile 32(M) x 128(N); each wave: 16x32 strip.
// A staged through LDS as f16; B fragments read from L2-resident packed f16.
// a_mode: 0 = A from memory; 1 = on-the-fly A[e][k] = P1[r(e),k]*P2[c(e),k]
// c_mode: 0 = store C (+bias,+C); 1 = fused message scatter:
//             atomicAdd(C[c(e)*FF+col], (acc+bias) * P1[r(e)*FF+col])
__global__ __launch_bounds__(256)
void k_gemm(const float* __restrict__ A, const _Float16* __restrict__ Wp,
            const float* __restrict__ bias, float* __restrict__ C,
            const int* __restrict__ ei, const float* __restrict__ P1,
            const float* __restrict__ P2,
            int M, int Kpad, int Nout, int pre_relu, int add_c,
            int a_mode, int c_mode)
{
    __shared__ _Float16 As[32][40];          // row stride 80B (16B aligned)
    const int tid  = threadIdx.x;
    const int bm   = blockIdx.x * 32;
    const int bn   = blockIdx.y * 128;
    const int wave = tid >> 5, lane = tid & 31;
    const int wm   = wave >> 2, wn = wave & 3;
    const int sr   = tid >> 3;               // staging row 0..31
    const int sc   = (tid & 7) * 4;          // staging col {0,4,...,28}
    const int mr   = lane & 15;
    const int kb   = (lane >> 4) * 8;        // 0 or 8 (CDNA5 16-bit frag split)

    int erow = bm + sr; if (erow > M - 1) erow = M - 1;   // clamp: row-local effect only
    const float* ap0 = A  + (size_t)erow * Kpad + sc;     // a_mode 0
    const float* yp0 = nullptr; const float* mp0 = nullptr;
    if (a_mode) {                                          // a_mode 1: dG on the fly
        int r = ei[erow], c = ei[NE + erow];
        yp0 = P1 + (size_t)r * FF + sc;
        mp0 = P2 + (size_t)c * FF + sc;
    }
    const _Float16* w0 = Wp + (size_t)(bn + wn*32 + mr) * Kpad + kb;
    const _Float16* w1 = w0 + (size_t)16 * Kpad;

    v8f acc0 = {0.f,0.f,0.f,0.f,0.f,0.f,0.f,0.f};
    v8f acc1 = {0.f,0.f,0.f,0.f,0.f,0.f,0.f,0.f};

    for (int k0 = 0; k0 < Kpad; k0 += 32) {
        // ---- stage A tile: vector loads + packed cvt + one ds b64 ----
        v4f v;
        if (!a_mode) {
            v = *(const v4f*)(ap0 + k0);
            __builtin_prefetch(ap0 + k0 + 32, 0, 1);
        } else {
            v4f a = *(const v4f*)(yp0 + k0);
            v4f b = *(const v4f*)(mp0 + k0);
            v[0]=a[0]*b[0]; v[1]=a[1]*b[1]; v[2]=a[2]*b[2]; v[3]=a[3]*b[3];
        }
        if (pre_relu) {
            v[0]=fmaxf(v[0],0.f); v[1]=fmaxf(v[1],0.f);
            v[2]=fmaxf(v[2],0.f); v[3]=fmaxf(v[3],0.f);
        }
        v4h hv;
        hv[0]=(_Float16)v[0]; hv[1]=(_Float16)v[1];
        hv[2]=(_Float16)v[2]; hv[3]=(_Float16)v[3];
        *(v4h*)&As[sr][sc] = hv;
        __syncthreads();

        // ---- A fragment: two ds_load_b128, CDNA5 16-bit layout ----
        v8h alo = *(const v8h*)&As[wm*16 + mr][kb];
        v8h ahi = *(const v8h*)&As[wm*16 + mr][kb + 16];
        v16h af = __builtin_shufflevector(alo, ahi,
                   0,1,2,3,4,5,6,7,8,9,10,11,12,13,14,15);
        // ---- B fragments: two global b128 each, straight from L2 ----
        v8h b0l = *(const v8h*)(w0 + k0);
        v8h b0h = *(const v8h*)(w0 + k0 + 16);
        v16h bf0 = __builtin_shufflevector(b0l, b0h,
                   0,1,2,3,4,5,6,7,8,9,10,11,12,13,14,15);
        v8h b1l = *(const v8h*)(w1 + k0);
        v8h b1h = *(const v8h*)(w1 + k0 + 16);
        v16h bf1 = __builtin_shufflevector(b1l, b1h,
                   0,1,2,3,4,5,6,7,8,9,10,11,12,13,14,15);

        acc0 = __builtin_amdgcn_wmma_f32_16x16x32_f16(
                   false, af, false, bf0, (short)0, acc0, false, false);
        acc1 = __builtin_amdgcn_wmma_f32_16x16x32_f16(
                   false, af, false, bf1, (short)0, acc1, false, false);
        __syncthreads();
    }

    const int col0  = bn + wn*32 + mr;
    const int col1  = col0 + 16;
    const int rbase = bm + wm*16 + ((lane < 16) ? 0 : 8);
    const float ba0 = bias ? bias[col0] : 0.f;
    const float ba1 = bias ? bias[col1] : 0.f;
    if (c_mode == 0) {
        #pragma unroll
        for (int j = 0; j < 8; ++j) {
            int row = rbase + j;
            if (row < M) {
                size_t o0 = (size_t)row * Nout + col0;
                size_t o1 = (size_t)row * Nout + col1;
                float v0 = acc0[j] + ba0;
                float v1 = acc1[j] + ba1;
                if (add_c) { v0 += C[o0]; v1 += C[o1]; }
                C[o0] = v0; C[o1] = v1;
            }
        }
    } else {
        // fused message: g never hits memory
        #pragma unroll
        for (int j = 0; j < 8; ++j) {
            int e = rbase + j;
            if (e < M) {
                int r = ei[e], cdx = ei[NE + e];
                float g0 = acc0[j] + ba0;
                float g1 = acc1[j] + ba1;
                atomicAdd(&C[(size_t)cdx*FF + col0], g0 * P1[(size_t)r*FF + col0]);
                atomicAdd(&C[(size_t)cdx*FF + col1], g1 * P1[(size_t)r*FF + col1]);
            }
        }
    }
}

// ======================= elementwise =========================
__global__ void k_zero(float* p, int n){ int i=blockIdx.x*256+threadIdx.x; if(i<n) p[i]=0.f; }
__global__ void k_copy(float* d, const float* s, int n){ int i=blockIdx.x*256+threadIdx.x; if(i<n) d[i]=s[i]; }
__global__ void k_add (float* d, const float* s, int n){ int i=blockIdx.x*256+threadIdx.x; if(i<n) d[i]+=s[i]; }
__global__ void k_scale_u(float* d, const float* s, const float* u, int n){
    int i=blockIdx.x*256+threadIdx.x; if(i<n) d[i]=u[i&(FF-1)]*s[i];
}
__global__ void k_mask(float* d, const float* s, const float* x, int n){
    int i=blockIdx.x*256+threadIdx.x; if(i<n) d[i] = (x[i]>0.f) ? s[i] : 0.f;
}
__global__ void k_mask_add(float* d, const float* s, const float* x, int n){
    int i=blockIdx.x*256+threadIdx.x; if(i<n && x[i]>0.f) d[i]+=s[i];
}
__global__ void k_bcast_mask(float* d, const float* wv, const float* x, int n){
    int i=blockIdx.x*256+threadIdx.x; if(i<n) d[i] = (x[i]>0.f) ? wv[i&(FF-1)] : 0.f;
}
__global__ void k_embed(float* h, const float* emb, const int* z, int n){
    int i=blockIdx.x*256+threadIdx.x;
    if(i<n){ int a=i>>8, f=i&(FF-1); h[i]=emb[(size_t)z[a]*FF+f]; }
}

// ======================= edge kernels =========================
__global__ void k_edge_rbf(const float* __restrict__ pos, const int* __restrict__ ei,
                           const float* __restrict__ centers, const float* __restrict__ widths,
                           float* __restrict__ EA)
{
    int e = blockIdx.x, k = threadIdx.x;          // 128 threads/block
    int r = ei[e], c = ei[NE + e];
    float dx = pos[r*3+0]-pos[c*3+0];
    float dy = pos[r*3+1]-pos[c*3+1];
    float dz = pos[r*3+2]-pos[c*3+2];
    float d  = sqrtf(dx*dx+dy*dy+dz*dz);
    float x  = d * (1.0f/RCUT);
    float cv = 0.f;
    if (x < 1.f) {
        float x3=x*x*x, x4=x3*x, x5=x4*x;
        cv = 1.f - 6.f*x5 + 15.f*x4 - 10.f*x3;
    }
    float val = 0.f;
    if (k < DF) {
        float t = __expf(-d);
        float df = t - centers[k];
        val = cv * __expf(-widths[k]*df*df);
    }
    EA[(size_t)e*DFP + k] = val;
}

// dY[r,f] += G[e,f] * dM[c,f]   (dG handled on the fly inside k_gemm a_mode=1)
__global__ void k_msg_bwd(const int* __restrict__ ei, const float* __restrict__ G,
                          const float* __restrict__ dM, float* __restrict__ dY)
{
    int e = blockIdx.x, f = threadIdx.x;
    int r = ei[e], c = ei[NE + e];
    float v = dM[(size_t)c*FF+f];
    atomicAdd(&dY[(size_t)r*FF+f], G[(size_t)e*FF+f] * v);
}

__global__ void k_out_dot(const float* __restrict__ O2, const float* __restrict__ wv,
                          const float* __restrict__ bv, float* __restrict__ Eacc)
{
    int n = blockIdx.x*256+threadIdx.x; if(n>=NA) return;
    float s = bv[0];
    for (int f=0; f<FF; ++f) {
        float v = O2[(size_t)n*FF+f];
        if (v>0.f) s += v*wv[f];
    }
    Eacc[n] += s;
}

__global__ void k_seg_energy(float* __restrict__ out, const int* __restrict__ batch,
                             const float* __restrict__ Eacc)
{
    int n = blockIdx.x*256+threadIdx.x; if(n>=NA) return;
    atomicAdd(&out[batch[n]], Eacc[n]);
}

__global__ void k_force(const float* __restrict__ pos, const int* __restrict__ ei,
                        const float* __restrict__ centers, const float* __restrict__ widths,
                        const float* __restrict__ dEA, float* __restrict__ force)
{
    int e = blockIdx.x*256+threadIdx.x; if(e>=NE) return;
    int r = ei[e], c = ei[NE + e];
    float dx = pos[r*3+0]-pos[c*3+0];
    float dy = pos[r*3+1]-pos[c*3+1];
    float dz = pos[r*3+2]-pos[c*3+2];
    float d  = sqrtf(dx*dx+dy*dy+dz*dz);
    if (d < 1e-10f) return;
    float x  = d*(1.0f/RCUT);
    float cv = 0.f, dcv = 0.f;
    if (x < 1.f) {
        float x2=x*x, x3=x2*x, x4=x3*x, x5=x4*x;
        cv  = 1.f - 6.f*x5 + 15.f*x4 - 10.f*x3;
        dcv = (1.0f/RCUT)*(-30.f*x4 + 60.f*x3 - 30.f*x2);
    }
    float t = __expf(-d);
    float dd = 0.f;
    for (int k=0;k<DF;++k) {
        float df = t - centers[k];
        float rb = __expf(-widths[k]*df*df);
        float g  = dEA[(size_t)e*DFP+k];
        dd += g * (dcv*rb + cv*rb*(2.f*widths[k]*df*t));
    }
    float inv = 1.f/d;
    float ux=dx*inv, uy=dy*inv, uz=dz*inv;
    atomicAdd(&force[r*3+0], -dd*ux); atomicAdd(&force[r*3+1], -dd*uy); atomicAdd(&force[r*3+2], -dd*uz);
    atomicAdd(&force[c*3+0],  dd*ux); atomicAdd(&force[c*3+1],  dd*uy); atomicAdd(&force[c*3+2],  dd*uz);
}

// ======================= host orchestration =========================
extern "C" void kernel_launch(void* const* d_in, const int* in_sizes, int n_in,
                              void* d_out, int out_size, void* d_ws, size_t ws_size,
                              hipStream_t stream)
{
    (void)in_sizes; (void)n_in; (void)out_size; (void)ws_size;
    const int*   z       = (const int*)  d_in[0];
    const float* pos     = (const float*)d_in[1];
    const int*   ei      = (const int*)  d_in[2];
    const int*   batch   = (const int*)  d_in[3];
    const float* centers = (const float*)d_in[4];
    const float* embed   = (const float*)d_in[5];
    const float* widths  = (const float*)d_in[72];

    struct Inter { const float *d2f_b,*d2f_w,*dense_b,*dense_w,*di_b,*di_w,
                   *res_db[2],*res_dw[2],*res_rb[2],*res_rw[2],
                   *ra_db[2],*ra_dw[2],*ra_rb[2],*ra_rw[2],*u; };
    struct OutB  { const float *ob,*ow,*res_db[2],*res_dw[2],*res_rb[2],*res_rw[2]; };
    Inter ip[2]; OutB op[2];
    int idx = 6;
    auto F32 = [&](int i){ return (const float*)d_in[i]; };
    for (int i=0;i<2;++i) {
        ip[i].d2f_b  = F32(idx++); ip[i].d2f_w  = F32(idx++);
        ip[i].dense_b= F32(idx++); ip[i].dense_w= F32(idx++);
        ip[i].di_b   = F32(idx++); ip[i].di_w   = F32(idx++);
        for (int r=0;r<2;++r){ ip[i].res_db[r]=F32(idx++); ip[i].res_dw[r]=F32(idx++);
                               ip[i].res_rb[r]=F32(idx++); ip[i].res_rw[r]=F32(idx++); }
        for (int r=0;r<2;++r){ ip[i].ra_db[r]=F32(idx++); ip[i].ra_dw[r]=F32(idx++);
                               ip[i].ra_rb[r]=F32(idx++); ip[i].ra_rw[r]=F32(idx++); }
        ip[i].u = F32(idx++);
    }
    for (int i=0;i<2;++i) {
        op[i].ob = F32(idx++); op[i].ow = F32(idx++);
        for (int r=0;r<2;++r){ op[i].res_db[r]=F32(idx++); op[i].res_dw[r]=F32(idx++);
                               op[i].res_rb[r]=F32(idx++); op[i].res_rw[r]=F32(idx++); }
    }

    // ---------------- workspace layout ----------------
    const size_t NF=(size_t)NA*FF, EFsz=(size_t)NE*FF, EDF=(size_t)NE*DFP;
    float* w = (float*)d_ws;
    auto take=[&](size_t n){ float* p=w; w+=n; return p; };
    float* EA  = take(EDF);  float* dEA = take(EDF);
    float* Gb  = take(EFsz);
    float* Hin[3]  = {take(NF),take(NF),take(NF)};
    float* Yb[2]   = {take(NF),take(NF)};
    float* M0b[2]  = {take(NF),take(NF)};
    float* M1b[2]  = {take(NF),take(NF)};
    float* M2b[2]  = {take(NF),take(NF)};
    float* Hmid[2] = {take(NF),take(NF)};
    float* A1b[2]  = {take(NF),take(NF)};
    float* O1b[2]  = {take(NF),take(NF)};
    float* O2b[2]  = {take(NF),take(NF)};
    float* D0 = take(NF); float* D1 = take(NF); float* OBt = take(NF);
    float* DM = take(NF); float* DY = take(NF);
    float* T1 = take(NF); float* T2 = take(NF);
    float* Eacc = take(NA);
    _Float16* Wpack = (_Float16*)take(32768 + 64);   // 65536 halves max (256x256)
    float* Eout = (float*)d_out;           // [64] energies
    float* Fout = Eout + NG;               // [NA*3] forces

    // ---------------- helpers ----------------
    auto gemm_x = [&](const float* A, const float* W, const float* bias, float* C,
                      const float* P1, const float* P2,
                      int M, int Kpad, int Nout, int k_real, int n_real, int ldw,
                      bool prelu, bool trans, bool addc, int a_mode, int c_mode){
        int tot = Nout * Kpad;
        k_pack_w<<<dim3((tot+255)/256),256,0,stream>>>(W, Wpack, Kpad, Nout,
                                                       k_real, n_real, ldw, trans?1:0);
        dim3 g((M+31)/32, Nout/128);
        k_gemm<<<g,256,0,stream>>>(A, Wpack, bias, C, ei, P1, P2, M, Kpad, Nout,
                                   prelu?1:0, addc?1:0, a_mode, c_mode);
    };
    auto gemm = [&](const float* A, const float* W, const float* bias, float* C,
                    int M, int Kpad, int Nout, int k_real, int n_real, int ldw,
                    bool prelu, bool trans, bool addc){
        gemm_x(A, W, bias, C, nullptr, nullptr, M, Kpad, Nout,
               k_real, n_real, ldw, prelu, trans, addc, 0, 0);
    };
    auto blks=[&](size_t n){ return dim3((unsigned)((n+255)/256)); };
    auto zero=[&](float* p, size_t n){ k_zero<<<blks(n),256,0,stream>>>(p,(int)n); };
    auto res_fwd=[&](const float* X, float* OUT, const float* Wd, const float* bd,
                     const float* Wr, const float* br){
        gemm(X, Wd, bd, T1, NA, FF, FF, FF, FF, FF, true, false, false);
        k_copy<<<blks(NF),256,0,stream>>>(OUT, X, (int)NF);
        gemm(T1, Wr, br, OUT, NA, FF, FF, FF, FF, FF, false, false, true);
    };
    auto res_bwd=[&](float* D, const float* X, const float* Wd, const float* Wr){
        gemm(D,  Wr, nullptr, T1, NA, FF, FF, FF, FF, FF, false, true, false);
        gemm(T1, Wd, nullptr, T2, NA, FF, FF, FF, FF, FF, false, true, false);
        k_mask_add<<<blks(NF),256,0,stream>>>(D, T2, X, (int)NF);
    };

    // ======================= forward =======================
    k_embed<<<blks(NF),256,0,stream>>>(Hin[0], embed, z, (int)NF);
    k_edge_rbf<<<NE,DFP,0,stream>>>(pos, ei, centers, widths, EA);
    zero(Eacc, NA);

    for (int i=0;i<2;++i) {
        gemm(Hin[i], ip[i].di_w, ip[i].di_b, Yb[i], NA, FF, FF, FF, FF, FF, true, false, false);
        // fused: m[col] += (EA@Wd2f + b) * y[row]   -- g never materialized
        zero(M0b[i], NF);
        gemm_x(EA, ip[i].d2f_w, ip[i].d2f_b, M0b[i], Yb[i], nullptr,
               NE, DFP, FF, DF, FF, FF, false, false, false, 0, 1);
        res_fwd(M0b[i], M1b[i], ip[i].res_dw[0], ip[i].res_db[0], ip[i].res_rw[0], ip[i].res_rb[0]);
        res_fwd(M1b[i], M2b[i], ip[i].res_dw[1], ip[i].res_db[1], ip[i].res_rw[1], ip[i].res_rb[1]);
        k_scale_u<<<blks(NF),256,0,stream>>>(Hmid[i], Hin[i], ip[i].u, (int)NF);
        gemm(M2b[i], ip[i].dense_w, ip[i].dense_b, Hmid[i], NA, FF, FF, FF, FF, FF, true, false, true);
        res_fwd(Hmid[i], A1b[i],   ip[i].ra_dw[0], ip[i].ra_db[0], ip[i].ra_rw[0], ip[i].ra_rb[0]);
        res_fwd(A1b[i],  Hin[i+1], ip[i].ra_dw[1], ip[i].ra_db[1], ip[i].ra_rw[1], ip[i].ra_rb[1]);
        res_fwd(Hin[i+1], O1b[i], op[i].res_dw[0], op[i].res_db[0], op[i].res_rw[0], op[i].res_rb[0]);
        res_fwd(O1b[i],   O2b[i], op[i].res_dw[1], op[i].res_db[1], op[i].res_rw[1], op[i].res_rb[1]);
        k_out_dot<<<blks(NA),256,0,stream>>>(O2b[i], op[i].ow, op[i].ob, Eacc);
    }
    zero(Eout, (size_t)NG + (size_t)NA*3);
    k_seg_energy<<<blks(NA),256,0,stream>>>(Eout, batch, Eacc);

    // ======================= backward (forces) =======================
    zero(dEA, EDF);
    float* dh = D0; float* dnx = D1;
    zero(dh, NF);
    for (int i=1;i>=0;--i) {
        // output block: dE/dEacc == 1 everywhere
        k_bcast_mask<<<blks(NF),256,0,stream>>>(OBt, op[i].ow, O2b[i], (int)NF);
        res_bwd(OBt, O1b[i],   op[i].res_dw[1], op[i].res_rw[1]);
        res_bwd(OBt, Hin[i+1], op[i].res_dw[0], op[i].res_rw[0]);
        k_add<<<blks(NF),256,0,stream>>>(dh, OBt, (int)NF);
        // atomic residuals
        res_bwd(dh, A1b[i],  ip[i].ra_dw[1], ip[i].ra_rw[1]);
        res_bwd(dh, Hmid[i], ip[i].ra_dw[0], ip[i].ra_rw[0]);
        // h-update: dmrelu = dh @ Wdense^T masked by M2; dHin partial = u*dh
        gemm(dh, ip[i].dense_w, nullptr, T1, NA, FF, FF, FF, FF, FF, false, true, false);
        k_mask<<<blks(NF),256,0,stream>>>(DM, T1, M2b[i], (int)NF);
        k_scale_u<<<blks(NF),256,0,stream>>>(dnx, dh, ip[i].u, (int)NF);
        // message residuals
        res_bwd(DM, M1b[i], ip[i].res_dw[1], ip[i].res_rw[1]);
        res_bwd(DM, M0b[i], ip[i].res_dw[0], ip[i].res_rw[0]);
        // recompute G (needed for dY), then message adjoint
        gemm(EA, ip[i].d2f_w, ip[i].d2f_b, Gb, NE, DFP, FF, DF, FF, FF, false, false, false);
        zero(DY, NF);
        k_msg_bwd<<<NE,FF,0,stream>>>(ei, Gb, DM, DY);
        // dEA += dG @ Wd2f^T with dG = Y[row]*dM[col] generated in-staging
        gemm_x(nullptr, ip[i].d2f_w, nullptr, dEA, Yb[i], DM,
               NE, FF, DFP, FF, DF, FF, false, true, true, 1, 0);
        // dHin += (dy @ Wdi^T) masked by Hin
        gemm(DY, ip[i].di_w, nullptr, T1, NA, FF, FF, FF, FF, FF, false, true, false);
        k_mask_add<<<blks(NF),256,0,stream>>>(dnx, T1, Hin[i], (int)NF);
        // swap
        float* tmp = dh; dh = dnx; dnx = tmp;
    }
    // reduce dEA -> per-edge dE/dd -> forces
    k_force<<<blks(NE),256,0,stream>>>(pos, ei, centers, widths, dEA, Fout);
}
